// ONLSTMCell_15367392985657
// MI455X (gfx1250) — matrix operand order repaired and checked by reading, not compiled, not run on hardware
//
#include <hip/hip_runtime.h>

#define INPUT_SIZE 1024
#define HIDDEN     1024
#define NCHUNK     64
#define GATE_W     4224      // 4*HIDDEN + 2*NCHUNK
#define K2         2048      // INPUT_SIZE + HIDDEN
#define BROWS      16384

#define BM         64        // rows per workgroup
#define ASTRIDE    2056      // padded bf16 elements per A row in LDS (bank-spread)
#define CCSTRIDE   132       // padded floats per row of phase-1 staging

typedef __attribute__((ext_vector_type(16))) __bf16 v16bf;
typedef __attribute__((ext_vector_type(8)))  float  v8f;

static __device__ __forceinline__ unsigned short f2bf(float f) {
  unsigned int u = __float_as_uint(f);
  unsigned int r = 0x7FFFu + ((u >> 16) & 1u);   // round-to-nearest-even
  return (unsigned short)((u + r) >> 16);
}

// A fragment (16x32 bf16): lane holds 16 consecutive K [base..base+15]
static __device__ __forceinline__ v16bf loadA(const unsigned short* p) {
  v16bf v;
  ((uint4*)&v)[0] = *(const uint4*)(p);
  ((uint4*)&v)[1] = *(const uint4*)(p + 8);
  return v;
}
// B fragment (32x16 bf16): lane holds 16 consecutive K of its column
static __device__ __forceinline__ v16bf loadB(const unsigned short* p) {
  v16bf v;
  ((uint4*)&v)[0] = *(const uint4*)(p);
  ((uint4*)&v)[1] = *(const uint4*)(p + 8);
  return v;
}

static __device__ __forceinline__ float sigmf(float x) {
  return 1.0f / (1.0f + __expf(-x));
}
static __device__ __forceinline__ float tanhf_fast(float x) {
  float e = __expf(-2.0f * fabsf(x));
  float t = (1.0f - e) / (1.0f + e);
  return copysignf(t, x);
}

// ---------------- prep: bias fold ----------------
__global__ void onlstm_prep_bias(const float* __restrict__ b_ih,
                                 const float* __restrict__ b_hh,
                                 float* __restrict__ biasWS) {
  int i = blockIdx.x * blockDim.x + threadIdx.x;
  if (i < GATE_W) biasWS[i] = b_ih[i] + b_hh[i];
}

// ---------------- prep: W transpose + f32->bf16 ----------------
// Wt[c][k] = (k<1024 ? W_ih[k][c] : W_hh[k-1024][c]) as bf16 (column-major weights)
__global__ void onlstm_prep_wt(const float* __restrict__ W_ih,
                               const float* __restrict__ W_hh,
                               unsigned short* __restrict__ Wt) {
  __shared__ float tile[32][33];
  const int c0 = blockIdx.x * 32, k0 = blockIdx.y * 32;
  const int tx = threadIdx.x, ty = threadIdx.y;
  for (int i = ty; i < 32; i += 8) {
    int k = k0 + i, c = c0 + tx;
    float v = (k < INPUT_SIZE) ? W_ih[(size_t)k * GATE_W + c]
                               : W_hh[(size_t)(k - INPUT_SIZE) * GATE_W + c];
    tile[i][tx] = v;
  }
  __syncthreads();
  for (int i = ty; i < 32; i += 8) {
    int c = c0 + i, k = k0 + tx;
    Wt[(size_t)c * K2 + k] = f2bf(tile[tx][i]);
  }
}

// ---------------- fused main kernel ----------------
__global__ __launch_bounds__(512, 1) void onlstm_main(
    const float* __restrict__ x, const float* __restrict__ hx,
    const float* __restrict__ cx, const unsigned short* __restrict__ Wt,
    const float* __restrict__ biasWS, float* __restrict__ out)
{
  __shared__ __align__(16) unsigned short Als[BM * ASTRIDE];  // ~257 KB, A-block resident
  __shared__ float cc[BM * CCSTRIDE];                         // ~33 KB, cin/cforget staging

  const int tid  = threadIdx.x;
  const int wave = tid >> 5;
  const int lane = tid & 31;
  const int ln   = lane & 15;
  const int lh   = lane >> 4;                 // 0 or 1 (half-wave)
  const size_t rowBase = (size_t)blockIdx.x * BM;

  // ---- stage A = [x | hx] for 64 rows, f32 -> bf16, into LDS (loaded once) ----
  for (int v = tid; v < BM * (K2 / 4); v += 512) {
    int m  = v >> 9;                         // K2/4 == 512 float4 per row
    int p4 = (v & 511) * 4;
    size_t grow = rowBase + (size_t)m;
    const float* src = (p4 < INPUT_SIZE) ? (x + grow * INPUT_SIZE + p4)
                                         : (hx + grow * HIDDEN + (p4 - INPUT_SIZE));
    float4 f = *(const float4*)src;
    ushort4 h;
    h.x = f2bf(f.x); h.y = f2bf(f.y); h.z = f2bf(f.z); h.w = f2bf(f.w);
    *(ushort4*)&Als[m * ASTRIDE + p4] = h;
  }
  __syncthreads();

  // ---- Phase 1: gate columns 0..127 (cingate | cforgetgate), B double-buffered ----
  {
    const int n   = wave & 7;                // N-tile 0..7  (cols 0..127)
    const int mh  = wave >> 3;               // handles m-tiles {2mh, 2mh+1}
    const int col = n * 16 + ln;
    const unsigned short* bp  = Wt + (size_t)col * K2 + lh * 16;
    const unsigned short* a0p = &Als[(mh * 32 + 0  + ln) * ASTRIDE + lh * 16];
    const unsigned short* a1p = &Als[(mh * 32 + 16 + ln) * ASTRIDE + lh * 16];
    v8f acc0 = {}, acc1 = {};
    v16bf bfA = loadB(bp);
    #pragma unroll 1
    for (int k0 = 0; k0 < K2; k0 += 64) {
      v16bf bfB = loadB(bp + k0 + 32);
      v16bf a0 = loadA(a0p + k0);
      v16bf a1 = loadA(a1p + k0);
      acc0 = __builtin_amdgcn_wmma_f32_16x16x32_bf16(false, a0, false, bfA, (short)0, acc0, false, false);
      acc1 = __builtin_amdgcn_wmma_f32_16x16x32_bf16(false, a1, false, bfA, (short)0, acc1, false, false);
      bfA = loadB(bp + k0 + 64);             // last iter overshoots into the next Wt column (still in Wt): harmless
      v16bf a2 = loadA(a0p + k0 + 32);
      v16bf a3 = loadA(a1p + k0 + 32);
      acc0 = __builtin_amdgcn_wmma_f32_16x16x32_bf16(false, a2, false, bfB, (short)0, acc0, false, false);
      acc1 = __builtin_amdgcn_wmma_f32_16x16x32_bf16(false, a3, false, bfB, (short)0, acc1, false, false);
    }
    float bv = biasWS[col];
    #pragma unroll
    for (int r = 0; r < 8; ++r) {
      int lr0 = mh * 32 + 0  + r + lh * 8;
      int lr1 = mh * 32 + 16 + r + lh * 8;
      cc[lr0 * CCSTRIDE + col] = acc0[r] + bv;
      cc[lr1 * CCSTRIDE + col] = acc1[r] + bv;
    }
  }
  __syncthreads();

  // ---- cummax: per-row softmax + cumsum over 64 chunk gates ----
  if (tid < 128) {
    int row = tid >> 1, half = tid & 1;
    float* base = &cc[row * CCSTRIDE + half * 64];
    float mx = -3.0e38f;
    for (int j = 0; j < 64; ++j) mx = fmaxf(mx, base[j]);
    float s = 0.0f;
    for (int j = 0; j < 64; ++j) s += __expf(base[j] - mx);
    float inv = 1.0f / s;
    float run = 0.0f;
    for (int j = 0; j < 64; ++j) {
      run += __expf(base[j] - mx);
      float cm = run * inv;
      base[j] = half ? cm : (1.0f - cm);   // half0 -> cin = 1-cummax, half1 -> cforget
    }
  }
  __syncthreads();

  // ---- Phase 2: out/cell/in/forget planes, fused epilogue ----
  for (int grp = 0; grp < 4; ++grp) {
    const int h0 = grp * 256 + wave * 16;   // this wave's hidden-column base
    const int j  = h0 >> 4;                 // chunk index (tile is chunk-aligned)

    const unsigned short* bp[4];
    #pragma unroll
    for (int g = 0; g < 4; ++g)
      bp[g] = Wt + (size_t)(128 + g * HIDDEN + h0 + ln) * K2 + lh * 16;
    const unsigned short* ap[4];
    #pragma unroll
    for (int m = 0; m < 4; ++m)
      ap[m] = &Als[(m * 16 + ln) * ASTRIDE + lh * 16];

    v8f acc[16] = {};                       // [m_tile][gate]

    v16bf bA[4], bB[4], aF[4];
    #pragma unroll
    for (int g = 0; g < 4; ++g) bA[g] = loadB(bp[g]);

    #pragma unroll 1
    for (int k0 = 0; k0 < K2; k0 += 64) {
      // prefetch B for the second half-step into the other buffer (no WAR with bA)
      #pragma unroll
      for (int g = 0; g < 4; ++g) bB[g] = loadB(bp[g] + k0 + 32);
      #pragma unroll
      for (int m = 0; m < 4; ++m) aF[m] = loadA(ap[m] + k0);
      #pragma unroll
      for (int m = 0; m < 4; ++m) {
        #pragma unroll
        for (int g = 0; g < 4; ++g)
          acc[m * 4 + g] = __builtin_amdgcn_wmma_f32_16x16x32_bf16(
              false, aF[m], false, bA[g], (short)0, acc[m * 4 + g], false, false);
      }
      // prefetch B for next iteration's first half-step (overshoot on last iter is harmless)
      #pragma unroll
      for (int g = 0; g < 4; ++g) bA[g] = loadB(bp[g] + k0 + 64);
      #pragma unroll
      for (int m = 0; m < 4; ++m) aF[m] = loadA(ap[m] + k0 + 32);
      #pragma unroll
      for (int m = 0; m < 4; ++m) {
        #pragma unroll
        for (int g = 0; g < 4; ++g)
          acc[m * 4 + g] = __builtin_amdgcn_wmma_f32_16x16x32_bf16(
              false, aF[m], false, bB[g], (short)0, acc[m * 4 + g], false, false);
      }
    }

    // prefetch next group's B stream while the epilogue runs (global_prefetch_b8)
    if (grp < 3) {
      const int h0n = (grp + 1) * 256 + wave * 16;
      #pragma unroll
      for (int g = 0; g < 4; ++g)
        __builtin_prefetch((const void*)(Wt + (size_t)(128 + g * HIDDEN + h0n + ln) * K2), 0, 0);
    }

    float bv[4];
    #pragma unroll
    for (int g = 0; g < 4; ++g) bv[g] = biasWS[128 + g * HIDDEN + h0 + ln];

    #pragma unroll
    for (int m = 0; m < 4; ++m) {
      #pragma unroll
      for (int r = 0; r < 8; ++r) {
        int lrow = m * 16 + r + lh * 8;
        float cinv = cc[lrow * CCSTRIDE + j];
        float cfv  = cc[lrow * CCSTRIDE + 64 + j];
        float og = acc[m * 4 + 0][r] + bv[0];
        float cg = acc[m * 4 + 1][r] + bv[1];
        float ig = acc[m * 4 + 2][r] + bv[2];
        float fg = acc[m * 4 + 3][r] + bv[3];
        float ov   = cfv * cinv;
        float forg = sigmf(fg) * ov + (cfv - ov);
        float ing  = sigmf(ig) * ov + (cinv - ov);
        float cell = tanhf_fast(cg);
        float outg = sigmf(og);
        size_t idx = (rowBase + (size_t)lrow) * HIDDEN + h0 + ln;
        float cyv = forg * cx[idx] + ing * cell;
        out[idx] = outg * tanhf_fast(cyv);                 // hy
        out[(size_t)BROWS * HIDDEN + idx] = cyv;           // cy
      }
    }
  }
}

extern "C" void kernel_launch(void* const* d_in, const int* in_sizes, int n_in,
                              void* d_out, int out_size, void* d_ws, size_t ws_size,
                              hipStream_t stream) {
  const float* x    = (const float*)d_in[0];
  const float* hx   = (const float*)d_in[1];
  const float* cx   = (const float*)d_in[2];
  const float* W_ih = (const float*)d_in[3];
  const float* b_ih = (const float*)d_in[4];
  const float* W_hh = (const float*)d_in[5];
  const float* b_hh = (const float*)d_in[6];
  float* out = (float*)d_out;

  unsigned short* Wt = (unsigned short*)d_ws;                 // 4224*2048 bf16 = 16.5 MB
  float* biasWS = (float*)((char*)d_ws + (size_t)GATE_W * K2 * sizeof(unsigned short));

  onlstm_prep_wt<<<dim3(GATE_W / 32, K2 / 32), dim3(32, 8), 0, stream>>>(W_ih, W_hh, Wt);
  onlstm_prep_bias<<<dim3((GATE_W + 255) / 256), dim3(256), 0, stream>>>(b_ih, b_hh, biasWS);
  onlstm_main<<<dim3(BROWS / BM), dim3(512), 0, stream>>>(x, hx, cx, Wt, biasWS, out);
}
